// Self_attention2_57466662420739
// MI455X (gfx1250) — compile-verified, matrix operand order checked by reading, not run on hardware
//
#include <hip/hip_runtime.h>
#include <stdint.h>

// out = (relu(x)+x) @ W2^T + b2   -- everything else in the reference is dead code.
// M=16384 (B*S), N=K=1024 (D). f16 WMMA, f32 accumulate.

#define D_DIM   1024
#define M_ROWS  16384
#define MBLK    128            // rows per workgroup
#define LDA     1032           // A LDS row stride in halves (padded: 2064B -> 4-bank skew)
#define BLDA    40             // B LDS row stride in halves (32 data + 8 pad via TDM pad)
#define KCH     32             // WMMA K
#define NPASSES 8              // 8 * 128 = 1024 N columns
#define NBUF    3              // triple-buffered B staging (one barrier per K-step)

typedef _Float16 v4h   __attribute__((ext_vector_type(4)));
typedef _Float16 v8h   __attribute__((ext_vector_type(8)));
typedef _Float16 v16h  __attribute__((ext_vector_type(16)));
typedef float    v8f   __attribute__((ext_vector_type(8)));
typedef float    f32x4 __attribute__((ext_vector_type(4)));
typedef unsigned int u32x4 __attribute__((ext_vector_type(4)));
typedef int      i32x4 __attribute__((ext_vector_type(4)));
typedef int      i32x8 __attribute__((ext_vector_type(8)));

union V16 { v16h v; v8h h[2]; };

// TDM: load a 2D tile (tile_dim0=32 halves contiguous, tile_dim1=128 rows,
// tensor row stride = 1024 halves) from global f16 W2 into LDS, padding the
// LDS destination by 4 DWORDs (16B) after every 16 DWORDs (one 64B row) so
// LDS rows land at stride BLDA=40 halves (bank-conflict-friendly).
__device__ __forceinline__ void tdm_load_tile(unsigned lds_byte_addr, const _Float16* gsrc) {
  unsigned long long ga = (unsigned long long)(uintptr_t)gsrc;
  u32x4 g0;
  g0[0] = 1u;                                                   // count=1, user descriptor
  g0[1] = lds_byte_addr;                                        // LDS dest (byte offset)
  g0[2] = (unsigned)ga;                                         // global_addr[31:0]
  g0[3] = (unsigned)((ga >> 32) & 0x01FFFFFFu) | 0x80000000u;   // addr[56:32] | type=2<<30
  i32x8 g1;
  g1[0] = (1 << 16)      // data_size = 1 -> 2 bytes
        | (1 << 20)      // pad_enable
        | (3 << 22)      // pad_interval code 3 -> every 16 DWORDs (64B row)
        | (3 << 25);     // pad_amount  code 3 -> 4 DWORDs (16B)
  g1[1] = (int)(((unsigned)D_DIM & 0xFFFFu) << 16);  // tensor_dim0[15:0] in bits [63:48]
  g1[2] = (int)(((unsigned)D_DIM & 0xFFFFu) << 16);  // dim0 hi=0; tensor_dim1[15:0] in [95:80]
  g1[3] = (int)(32u << 16);                          // dim1 hi=0; tile_dim0=32 in [127:112]
  g1[4] = 128;                                       // tile_dim1=128, tile_dim2=0
  g1[5] = D_DIM;                                     // tensor_dim0_stride lo32 = 1024
  g1[6] = 0;
  g1[7] = 0;
  i32x4 gz = {};
#if __clang_major__ >= 23
  i32x8 gz8 = {};
  __builtin_amdgcn_tensor_load_to_lds(g0, g1, gz, gz, gz8, 0);
#else
  __builtin_amdgcn_tensor_load_to_lds(g0, g1, gz, gz, 0);
#endif
}

extern "C" {

// W2 (f32, [N=1024][K=1024] row-major) -> f16, same layout, into workspace.
__global__ __launch_bounds__(256)
void convert_w2_f16(const float* __restrict__ w, _Float16* __restrict__ wh) {
  int i = blockIdx.x * blockDim.x + threadIdx.x;   // over 262144 float4s
  f32x4 f = __builtin_nontemporal_load(&((const f32x4*)w)[i]);
  v4h h;
  h[0] = (_Float16)f[0]; h[1] = (_Float16)f[1];
  h[2] = (_Float16)f[2]; h[3] = (_Float16)f[3];
  ((v4h*)wh)[i] = h;
}

__global__ __launch_bounds__(256)
void fused_residual_gemm(const float* __restrict__ x,
                         const _Float16* __restrict__ w2h,
                         const float* __restrict__ b2,
                         float* __restrict__ out) {
  extern __shared__ _Float16 smem[];
  _Float16* As = smem;                    // MBLK x LDA halves (f16 A tile, full K)
  _Float16* Bs = smem + MBLK * LDA;       // NBUF x 128 x BLDA halves (B staging)

  const int tid  = threadIdx.x;
  const int wave = tid >> 5;
  const int lane = tid & 31;
  const long rowBase = (long)blockIdx.x * MBLK;

  // ---- Stage A: x block -> relu(x)+x -> f16 -> LDS (x read exactly once) ----
  const f32x4* xg = (const f32x4*)(x + rowBase * D_DIM);
  for (int idx = tid; idx < MBLK * (D_DIM / 4); idx += 256) {
    int r  = idx / (D_DIM / 4);
    int c4 = idx % (D_DIM / 4);
    f32x4 f = __builtin_nontemporal_load(&xg[r * (D_DIM / 4) + c4]);
    v4h h;
    h[0] = (_Float16)(f[0] + fmaxf(f[0], 0.f));
    h[1] = (_Float16)(f[1] + fmaxf(f[1], 0.f));
    h[2] = (_Float16)(f[2] + fmaxf(f[2], 0.f));
    h[3] = (_Float16)(f[3] + fmaxf(f[3], 0.f));
    *(v4h*)(As + r * LDA + c4 * 4) = h;
  }
  __syncthreads();

  // Per-lane fragment coordinates (contiguous k order, used for both A and B):
  // A frag: lane 0-15 row M=lane, K 0..15; lane 16-31 same rows, K 16..31.
  const int mloc = wave * 16 + (lane & 15);
  const int kh   = (lane >> 4) * 16;
  const unsigned bsBase   = (unsigned)(uintptr_t)Bs;   // LDS aperture low bits = LDS offset
  const unsigned bufBytes = 128u * BLDA * 2u;

  for (int np = 0; np < NPASSES; ++np) {
    v8f acc[8];
    v8f zero = {};
#pragma unroll
    for (int t = 0; t < 8; ++t) acc[t] = zero;

    const _Float16* wsrc = w2h + (long)(np * 128) * D_DIM;
    if (wave == 0) tdm_load_tile(bsBase, wsrc);          // chunk 0 -> buf 0

    int bcur = 0;
#pragma unroll 1
    for (int ks = 0; ks < D_DIM / KCH; ++ks) {
      // Triple-buffer pipeline, ONE barrier per iteration:
      //  - chunk ks+1 goes to buf (bcur+1)%3; its previous tenant (chunk ks-2)
      //    was consumed before the barrier of iteration ks-1, already passed.
      //  - s_wait_tensorcnt(1) before the barrier publishes chunk ks to all waves.
      if (wave == 0) {
        if (ks + 1 < D_DIM / KCH) {
          int bnext = (bcur + 1 == NBUF) ? 0 : bcur + 1;
          tdm_load_tile(bsBase + (unsigned)bnext * bufBytes, wsrc + (ks + 1) * KCH);
          __builtin_amdgcn_s_wait_tensorcnt(1);
        } else {
          __builtin_amdgcn_s_wait_tensorcnt(0);
        }
      }
      __syncthreads();

      const _Float16* bbuf = Bs + bcur * (128 * BLDA);
      V16 a;
      const v8h* ap = (const v8h*)(As + mloc * LDA + ks * KCH + kh);
      a.h[0] = ap[0]; a.h[1] = ap[1];

      // Preload all 8 B fragments so the 16 ds_load_b128 issue back-to-back
      // and LDS latency overlaps the WMMA burst (no per-tile dscnt=0 stalls).
      V16 bf[8];
      const v8h* bp = (const v8h*)(bbuf + (lane & 15) * BLDA + kh);
#pragma unroll
      for (int t = 0; t < 8; ++t) {
        bf[t].h[0] = bp[t * (16 * BLDA / 8)];
        bf[t].h[1] = bp[t * (16 * BLDA / 8) + 1];
      }
#pragma unroll
      for (int t = 0; t < 8; ++t)
        acc[t] = __builtin_amdgcn_wmma_f32_16x16x32_f16(
            false, a.v, false, bf[t].v, (short)0, acc[t], false, false);

      bcur = (bcur + 1 == NBUF) ? 0 : bcur + 1;
    }

    // ---- Epilogue: + b2, NT store. D layout: lanes 0-15 M=r, lanes 16-31 M=8+r ----
    const int mout = (int)rowBase + wave * 16 + (lane >> 4) * 8;
#pragma unroll
    for (int t = 0; t < 8; ++t) {
      int n = np * 128 + t * 16 + (lane & 15);
      float bias = b2[n];
#pragma unroll
      for (int r = 0; r < 8; ++r)
        __builtin_nontemporal_store(acc[t][r] + bias,
                                    out + (long)(mout + r) * D_DIM + n);
    }
  }
}

void kernel_launch(void* const* d_in, const int* in_sizes, int n_in,
                   void* d_out, int out_size, void* d_ws, size_t ws_size,
                   hipStream_t stream) {
  (void)in_sizes; (void)n_in; (void)out_size; (void)ws_size;
  const float* x  = (const float*)d_in[0];   // [B,S,D]
  const float* W2 = (const float*)d_in[9];   // [D,D]
  const float* b2 = (const float*)d_in[10];  // [D]
  float* out = (float*)d_out;
  _Float16* w2h = (_Float16*)d_ws;           // 2 MB f16 copy of W2

  convert_w2_f16<<<(D_DIM * D_DIM / 4) / 256, 256, 0, stream>>>(W2, w2h);

  size_t lds = (size_t)(MBLK * LDA + NBUF * 128 * BLDA) * sizeof(_Float16);  // ~288 KB
  (void)hipFuncSetAttribute((const void*)fused_residual_gemm,
                            hipFuncAttributeMaxDynamicSharedMemorySize, (int)lds);
  fused_residual_gemm<<<M_ROWS / MBLK, 256, lds, stream>>>(x, w2h, b2, out);
}

} // extern "C"
